// _MultiHeadAttentionCore_14886356648345
// MI455X (gfx1250) — matrix operand order repaired and checked by reading, not compiled, not run on hardware
//
#include <hip/hip_runtime.h>

typedef __attribute__((ext_vector_type(16))) __bf16        v16bf;
typedef __attribute__((ext_vector_type(8)))  float         v8f;
typedef __attribute__((ext_vector_type(4)))  unsigned int  v4u;
typedef __attribute__((ext_vector_type(8)))  int           v8i;
typedef __attribute__((ext_vector_type(4)))  int           v4i;

#define B_  4
#define S_  2048
#define D_  1024
#define H_  16
#define DH_ 64

#if defined(__has_builtin)
# if __has_builtin(__builtin_amdgcn_tensor_load_to_lds)
#  define HAS_TDM 1
# endif
#endif
#ifndef HAS_TDM
# define HAS_TDM 0
#endif

__device__ __forceinline__ v8f wmma_bf16(v16bf a, v16bf b, v8f c) {
    // D = A(16x32 bf16) x B(32x16 bf16) + C(16x16 f32)
    return __builtin_amdgcn_wmma_f32_16x16x32_bf16(
        /*neg_a=*/false, a, /*neg_b=*/false, b,
        /*c_mod=*/(short)0, c, /*reuse_a=*/false, /*reuse_b=*/false);
}

// A-matrix (16x32, bf16) fragment from an f32 row.
// ISA layout: lanes 0-15 -> a[0..7]=K0..7, a[8..15]=K16..23 ;
//             lanes 16-31 -> a[0..7]=K8..15, a[8..15]=K24..31.
__device__ __forceinline__ v16bf load_a_f32(const float* row, int half, int dof) {
    v16bf a;
    const float* p = row + dof;
#pragma unroll
    for (int i = 0; i < 8; ++i) a[i] = (__bf16)p[half * 8 + i];
#pragma unroll
    for (int i = 0; i < 8; ++i) a[8 + i] = (__bf16)p[16 + half * 8 + i];
    return a;
}

// B-matrix (32x16, bf16) fragment where B[k][n] = lane_row[dof + k].
// ISA layout: lanes 0-15 -> b[0..7]=K0..7, b[8..15]=K16..23 ; lanes 16-31 -> b[0..7]=K8..15, b[8..15]=K24..31 (per-lane col n).
__device__ __forceinline__ v16bf load_b_row(const float* lane_row, int half, int dof) {
    v16bf b;
#pragma unroll
    for (int i = 0; i < 16; ++i) b[i] = (__bf16)lane_row[dof + half * 8 + i + (i >> 3) * 8];
    return b;
}

// ---- Tensor Data Mover: 2-D f32 tile (tile_d0 x tile_d1) -> LDS ----------
__device__ __forceinline__ void tdm_load_2d(unsigned int lds_off, const float* gptr,
                                            unsigned int tile_d0, unsigned int tile_d1,
                                            unsigned int tdim0, unsigned int tdim1,
                                            unsigned int stride0) {
#if HAS_TDM
    unsigned long long ga = (unsigned long long)(uintptr_t)gptr;
    v4u g0 = { 1u,                                   // count=1 user descriptor
               lds_off,                              // lds_addr (bytes)
               (unsigned int)ga,                     // global_addr[31:0]
               ((unsigned int)(ga >> 32) & 0x1FFFFFFu) | (2u << 30) }; // [56:32]|type=2
    v8i g1 = { (int)(2u << 16),                                  // data_size = 4B
               (int)((tdim0 & 0xFFFFu) << 16),                   // tensor_dim0 lo16
               (int)((tdim0 >> 16) | ((tdim1 & 0xFFFFu) << 16)), // dim0 hi | dim1 lo
               (int)((tdim1 >> 16) | (tile_d0 << 16)),           // dim1 hi | tile_dim0
               (int)tile_d1,                                     // tile_dim1 (tile_dim2=0)
               (int)stride0,                                     // tensor_dim0_stride lo32
               0, 0 };
    v4i z4 = { 0, 0, 0, 0 };
# if __clang_major__ >= 23
    v8i z8 = { 0, 0, 0, 0, 0, 0, 0, 0 };
    __builtin_amdgcn_tensor_load_to_lds(g0, g1, z4, z4, z8, 0);
# else
    __builtin_amdgcn_tensor_load_to_lds(g0, g1, z4, z4, 0);
# endif
#else
    (void)lds_off; (void)gptr; (void)tile_d0; (void)tile_d1;
    (void)tdim0; (void)tdim1; (void)stride0;
#endif
}

#if HAS_TDM
#define TDM_WAIT_2() asm volatile("s_wait_tensorcnt 0x2" ::: "memory")
#define TDM_WAIT_0() asm volatile("s_wait_tensorcnt 0x0" ::: "memory")
#else
#define TDM_WAIT_2() asm volatile("" ::: "memory")
#define TDM_WAIT_0() asm volatile("" ::: "memory")
#endif

__global__ __launch_bounds__(32) void attn_fa_kernel(
        const float* __restrict__ Q, const float* __restrict__ K,
        const float* __restrict__ V, const unsigned char* __restrict__ kpm,
        float* __restrict__ X) {
    __shared__ float  KtL[2][32 * DH_];   // double-buffered K tile (32 keys x 64)
    __shared__ float  VtL[2][32 * DH_];   // double-buffered V tile
    __shared__ __bf16 Plds[16 * 32];      // P staging (C-layout -> A-layout)

    const int lane = threadIdx.x & 31;
    const int half = lane >> 4;      // 0: lanes 0-15, 1: lanes 16-31
    const int lm   = lane & 15;      // row (A) / col (B,C) index within tile

    const int t  = blockIdx.x;                   // one wave-tile per block
    const int qtiles = S_ / 16;
    const int qt = t % qtiles;
    const int bh = t / qtiles;
    const int h  = bh % H_;
    const int b  = bh / H_;
    const int q0 = qt * 16;

    const size_t bOff = (size_t)b * S_ * D_;
    const float* Qh = Q + bOff + h * DH_;
    const float* Kh = K + bOff + h * DH_;
    const float* Vh = V + bOff + h * DH_;

    const unsigned int koff[2] = { (unsigned int)(uintptr_t)&KtL[0][0],
                                   (unsigned int)(uintptr_t)&KtL[1][0] };
    const unsigned int voff[2] = { (unsigned int)(uintptr_t)&VtL[0][0],
                                   (unsigned int)(uintptr_t)&VtL[1][0] };

    // Q fragments (rows q0..q0+15, head-dim halves 0-31 / 32-63)
    const float* qrow = Qh + (size_t)(q0 + lm) * D_;
    v16bf aq0 = load_a_f32(qrow, half, 0);
    v16bf aq1 = load_a_f32(qrow, half, 32);

    v8f o0 = {}, o1 = {}, o2 = {}, o3 = {};
    float mrow[8], lrow[8];
#pragma unroll
    for (int r = 0; r < 8; ++r) { mrow[r] = -1e30f; lrow[r] = 0.0f; }

    const int nkt = (q0 + 16 + 31) >> 5;   // causal key-tile count (32 keys/tile)

    // ---- prologue: DMA tile 0 (K and V) into buffer 0 -------------------
#if HAS_TDM
    tdm_load_2d(koff[0], Kh, DH_, 32, D_, S_, D_);
    tdm_load_2d(voff[0], Vh, DH_, 32, D_, S_, D_);
#else
    for (int idx = lane; idx < 32 * DH_; idx += 32) {
        KtL[0][idx] = Kh[(size_t)(idx / DH_) * D_ + (idx % DH_)];
        VtL[0][idx] = Vh[(size_t)(idx / DH_) * D_ + (idx % DH_)];
    }
    asm volatile("s_wait_dscnt 0x0" ::: "memory");
#endif

    for (int kt = 0; kt < nkt; ++kt) {
        const int k0  = kt * 32;
        const int cur = kt & 1;

        // ---- prefetch next tile into the other buffer, then wait current
        if (kt + 1 < nkt) {
            const int nxt = (kt + 1) & 1;
#if HAS_TDM
            tdm_load_2d(koff[nxt], Kh + (size_t)(k0 + 32) * D_, DH_, 32, D_, S_, D_);
            tdm_load_2d(voff[nxt], Vh + (size_t)(k0 + 32) * D_, DH_, 32, D_, S_, D_);
#else
            for (int idx = lane; idx < 32 * DH_; idx += 32) {
                KtL[nxt][idx] = Kh[(size_t)(k0 + 32 + idx / DH_) * D_ + (idx % DH_)];
                VtL[nxt][idx] = Vh[(size_t)(k0 + 32 + idx / DH_) * D_ + (idx % DH_)];
            }
#endif
            TDM_WAIT_2();   // current K/V tile complete; next 2 still in flight
        } else {
            TDM_WAIT_0();
        }
#if !HAS_TDM
        asm volatile("s_wait_dscnt 0x0" ::: "memory");
#endif

        // ---- scores: two 16x16 C tiles, each K-dim split 0-31 / 32-63 ----
        v8f c0 = {}, c1 = {};
        {
            const float* kr0 = &KtL[cur][lm * DH_];          // key column n = lm
            c0 = wmma_bf16(aq0, load_b_row(kr0, half, 0),  c0);
            c0 = wmma_bf16(aq1, load_b_row(kr0, half, 32), c0);
            const float* kr1 = &KtL[cur][(16 + lm) * DH_];   // key column n+16
            c1 = wmma_bf16(aq0, load_b_row(kr1, half, 0),  c1);
            c1 = wmma_bf16(aq1, load_b_row(kr1, half, 32), c1);
        }

        // ---- masking (causal + key padding), scale 1/sqrt(64) ----
        const int kg0 = k0 + lm, kg1 = k0 + 16 + lm;
        const bool pad0 = kpm[(size_t)b * S_ + kg0] != 0;
        const bool pad1 = kpm[(size_t)b * S_ + kg1] != 0;
        float s0[8], s1[8];
#pragma unroll
        for (int r = 0; r < 8; ++r) {
            const int qg = q0 + r + 8 * half;
            s0[r] = (pad0 || kg0 > qg) ? -1e30f : c0[r] * 0.125f;
            s1[r] = (pad1 || kg1 > qg) ? -1e30f : c1[r] * 0.125f;
        }

        // ---- online softmax (row stats reduced over the 16-lane half-group)
#pragma unroll
        for (int r = 0; r < 8; ++r) {
            float tm = fmaxf(s0[r], s1[r]);
            tm = fmaxf(tm, __shfl_xor(tm, 1, 16));
            tm = fmaxf(tm, __shfl_xor(tm, 2, 16));
            tm = fmaxf(tm, __shfl_xor(tm, 4, 16));
            tm = fmaxf(tm, __shfl_xor(tm, 8, 16));
            const float nm = fmaxf(mrow[r], tm);
            const float p0 = __expf(s0[r] - nm);
            const float p1 = __expf(s1[r] - nm);
            float rs = p0 + p1;
            rs += __shfl_xor(rs, 1, 16);
            rs += __shfl_xor(rs, 2, 16);
            rs += __shfl_xor(rs, 4, 16);
            rs += __shfl_xor(rs, 8, 16);
            const float alpha = __expf(mrow[r] - nm);
            lrow[r] = lrow[r] * alpha + rs;
            mrow[r] = nm;
            o0[r] *= alpha; o1[r] *= alpha; o2[r] *= alpha; o3[r] *= alpha;
            s0[r] = p0; s1[r] = p1;
        }

        // ---- relayout P: C-layout -> A-layout via LDS tile ----
#pragma unroll
        for (int r = 0; r < 8; ++r) {
            const int row = r + 8 * half;
            Plds[row * 32 + lm]      = (__bf16)s0[r];
            Plds[row * 32 + 16 + lm] = (__bf16)s1[r];
        }
        asm volatile("s_wait_dscnt 0x0" ::: "memory");
        v16bf ap;
#pragma unroll
        for (int i = 0; i < 8; ++i) ap[i]     = Plds[lm * 32 + half * 8 + i];
#pragma unroll
        for (int i = 0; i < 8; ++i) ap[8 + i] = Plds[lm * 32 + 16 + half * 8 + i];

        // ---- O += P x V  (16x32 * 32x64 as 4 WMMAs), V from LDS ----
#pragma unroll
        for (int j = 0; j < 4; ++j) {
            v16bf bv;
#pragma unroll
            for (int i = 0; i < 16; ++i)
                bv[i] = (__bf16)VtL[cur][(half * 8 + i + (i >> 3) * 8) * DH_ + j * 16 + lm];
            v8f& oj = (j == 0) ? o0 : (j == 1) ? o1 : (j == 2) ? o2 : o3;
            oj = wmma_bf16(ap, bv, oj);
        }
    }

    // ---- epilogue: normalize rows and write attention output (f32) ----
#pragma unroll
    for (int r = 0; r < 8; ++r) {
        const float inv = 1.0f / lrow[r];
        const int row = q0 + r + 8 * half;
        float* xr = X + bOff + (size_t)row * D_ + h * DH_;
        xr[0 * 16 + lm] = o0[r] * inv;
        xr[1 * 16 + lm] = o1[r] * inv;
        xr[2 * 16 + lm] = o2[r] * inv;
        xr[3 * 16 + lm] = o3[r] * inv;
    }
}

// out[row, col] = sum_d X[row, d] * W[col, d] + bias[col]
__global__ __launch_bounds__(128) void proj_kernel(
        const float* __restrict__ X, const float* __restrict__ W,
        const float* __restrict__ bias, float* __restrict__ out) {
    const int lane = threadIdx.x & 31;
    const int w    = threadIdx.x >> 5;
    const int half = lane >> 4;
    const int lm   = lane & 15;

    const int t = blockIdx.x * 4 + w;
    const int ctiles = D_ / 16;
    const int ct = t % ctiles;
    const int rt = t / ctiles;
    const int row0 = rt * 16, col0 = ct * 16;

    const float* xrow = X + (size_t)(row0 + lm) * D_;   // A row per lane
    const float* wrow = W + (size_t)(col0 + lm) * D_;   // B col n = W row

    v8f c = {};
    for (int k = 0; k < D_; k += 32) {
        v16bf a = load_a_f32(xrow, half, k);
        v16bf bfrag = load_b_row(wrow, half, k);
        c = wmma_bf16(a, bfrag, c);
    }

    const float bv = bias[col0 + lm];
#pragma unroll
    for (int r = 0; r < 8; ++r) {
        const int row = row0 + r + 8 * half;
        out[(size_t)row * D_ + col0 + lm] = c[r] + bv;
    }
}

extern "C" void kernel_launch(void* const* d_in, const int* in_sizes, int n_in,
                              void* d_out, int out_size, void* d_ws, size_t ws_size,
                              hipStream_t stream) {
    const float*         Q    = (const float*)d_in[0];
    const float*         K    = (const float*)d_in[1];
    const float*         V    = (const float*)d_in[2];
    const unsigned char* kpm  = (const unsigned char*)d_in[3];  // bool mask [B,S]
    const float*         W    = (const float*)d_in[4];
    const float*         bias = (const float*)d_in[5];
    float* out = (float*)d_out;
    float* X   = (float*)d_ws;   // [B,S,D] f32 attention output (32 MiB)

    // attention: B*H*(S/16) = 8192 wave-tiles, 1 wave per block (TDM-fed LDS)
    attn_fa_kernel<<<B_ * H_ * (S_ / 16), 32, 0, stream>>>(Q, K, V, kpm, X);
    // projection: (B*S/16)*(D/16) = 32768 wave-tiles, 4 waves/block
    proj_kernel<<<((B_ * S_ / 16) * (D_ / 16)) / 4, 128, 0, stream>>>(X, W, bias, out);
}
